// LSTMModel_35673998361333
// MI455X (gfx1250) — matrix operand order repaired and checked by reading, not compiled, not run on hardware
//
#include <hip/hip_runtime.h>

// ---------------------------------------------------------------------------
// Fused 2-layer LSTM (B=1024, T=512, D=1, H=64) + final FC on MI455X (gfx1250).
//
// One workgroup (8 wave32 waves) per 16-row batch tile runs the whole
// sequential recurrence. Gate pre-activations via V_WMMA_F32_16X16X32_F16:
//   layer0: [16x64]  x [64x256]   -> 4 WMMA/wave/step
//   layer1: [16x128] x [128x256]  -> 8 WMMA/wave/step
// Weight B-fragments are time-invariant and live in VGPRs. Activations use
// branch-free hardware v_exp_f32 / v_rcp_f32 (wave-uniform gate constants),
// avoiding exec-mask divergence and IEEE-division slow paths.
// ---------------------------------------------------------------------------

typedef __attribute__((ext_vector_type(16))) _Float16 v16h;
typedef __attribute__((ext_vector_type(8)))  _Float16 v8h;
typedef __attribute__((ext_vector_type(4)))  _Float16 v4h;
typedef __attribute__((ext_vector_type(8)))  float    v8f;

#define B_  1024
#define T_  512
#define H_  64

#define XS_STRIDE 516   // floats per x row   (512 + 4 pad, 16B-aligned rows)
#define HC_STRIDE 136   // halfs  per h row   (128 + 8 pad, 16B-aligned rows)
#define GA_STRIDE 20    // floats per gate row (16 + 4 pad, 16B-aligned rows)

#define L2E_ 1.4426950408889634f

__device__ __forceinline__ float fast_rcp(float x)  { return __builtin_amdgcn_rcpf(x); }
__device__ __forceinline__ float fast_exp2(float x) { return __builtin_amdgcn_exp2f(x); }

// Unified gate activation, branch-free:
//   sigmoid(z) = rcp(1 + exp2(-L2E*z))            -> a = 1, k = -L2E
//   tanh(z)    = 2*rcp(1 + exp2(-2*L2E*z)) - 1    -> a = 2, k = -2*L2E
//   act(z)     = a * rcp(1 + exp2(k*z)) - (a - 1)
__device__ __forceinline__ float gate_act(float z, float a, float k, float am1) {
    return fmaf(a, fast_rcp(1.0f + fast_exp2(k * z)), -am1);
}
__device__ __forceinline__ float tanh_fast(float z) {
    return fmaf(2.0f, fast_rcp(1.0f + fast_exp2(-2.0f * L2E_ * z)), -1.0f);
}

// A-operand fragment (16x32 f16) for V_WMMA_F32_16X16X32_F16 from LDS.
// Lane l: M = l&15; lanes 0-15 hold K = kc*32 + {0..15},
//                   lanes 16-31 hold K = kc*32 + {16..31}.
__device__ __forceinline__ v16h load_afrag(const _Float16* hcat, int lm, int hi, int kc) {
    const _Float16* p = hcat + lm * HC_STRIDE + kc * 32 + hi * 16;
    v8h lo = *(const v8h*)p;          // K = kc*32 + hi*16     .. +7
    v8h hf = *(const v8h*)(p + 8);    // K = kc*32 + hi*16 + 8 .. +7
    return __builtin_shufflevector(lo, hf, 0,1,2,3,4,5,6,7,8,9,10,11,12,13,14,15);
}

// B-operand fragment (32x16 f16): lane l supplies column N = l&15,
// K = (l>=16 ? 16 : 0) + {0..15}. Column N of B == row N of the f32 weight
// matrix (gates = h @ W^T), so 16 contiguous floats, converted to f16.
__device__ __forceinline__ v16h cvt_bfrag(const float* p) {
    v16h r;
#pragma unroll
    for (int k = 0; k < 16; ++k) r[k] = (_Float16)p[k];
    return r;
}

__global__ __launch_bounds__(256) void lstm2_fused_wmma(
    const float* __restrict__ x,
    const float* __restrict__ W_ih0, const float* __restrict__ W_hh0,
    const float* __restrict__ b_ih0, const float* __restrict__ b_hh0,
    const float* __restrict__ W_ih1, const float* __restrict__ W_hh1,
    const float* __restrict__ b_ih1, const float* __restrict__ b_hh1,
    const float* __restrict__ W_fc,  const float* __restrict__ b_fc,
    float* __restrict__ out)
{
    __shared__ float    xs[16 * XS_STRIDE];    // staged x tile   (33024 B)
    __shared__ _Float16 hcat[16 * HC_STRIDE];  // [y0 | h1] f16   ( 4352 B)
    __shared__ float    gact[256 * GA_STRIDE]; // activated gates (20480 B)

    const int tid  = threadIdx.x;
    const int wv   = tid >> 5;      // wave 0..7 -> gate cols [32w, 32w+32)
    const int lane = tid & 31;
    const int hi   = lane >> 4;     // lane half
    const int lm   = lane & 15;
    const int b0   = blockIdx.x * 16;

    // ---- stage x tile (16 contiguous rows of 512 f32) into LDS ----
    {
        const float4* gx = (const float4*)(x + (size_t)b0 * T_);
#pragma unroll
        for (int i = 0; i < 8; ++i) {
            int idx = tid + i * 256;          // 2048 float4 total
            int m   = idx >> 7;               // row within tile
            int t4  = idx & 127;
            float4 v = gx[idx];
            *(float4*)&xs[m * XS_STRIDE + t4 * 4] = v;
        }
        for (int i = tid; i < 16 * HC_STRIDE; i += 256)
            hcat[i] = (_Float16)0.0f;         // h0 = h1 = 0
    }

    // ---- hoist time-invariant weights into VGPRs ----
    v16h B0[2][2];          // layer0: [n-tile][k-chunk], K = 64
    v16h B1[2][4];          // layer1: [n-tile][k-chunk], K = 128 = [W_ih1; W_hh1]
    float bias0r[2], bias1r[2], wihr[2];
#pragma unroll
    for (int nt = 0; nt < 2; ++nt) {
        const int n = wv * 32 + nt * 16 + lm;     // global gate column
        bias0r[nt] = b_ih0[n] + b_hh0[n];
        bias1r[nt] = b_ih1[n] + b_hh1[n];
        wihr[nt]   = W_ih0[n];                    // D == 1 -> rank-1 input term
#pragma unroll
        for (int kc = 0; kc < 2; ++kc)
            B0[nt][kc] = cvt_bfrag(W_hh0 + n * 64 + kc * 32 + hi * 16);
#pragma unroll
        for (int kc = 0; kc < 2; ++kc)
            B1[nt][kc] = cvt_bfrag(W_ih1 + n * 64 + kc * 32 + hi * 16);
#pragma unroll
        for (int kc = 0; kc < 2; ++kc)
            B1[nt][kc + 2] = cvt_bfrag(W_hh1 + n * 64 + kc * 32 + hi * 16);
    }

    // Wave-uniform activation constants (gate order i,f,g,o -> g uses tanh).
    const bool  tanh_gate = ((wv >> 1) == 2);
    const float act_a   = tanh_gate ? 2.0f : 1.0f;
    const float act_k   = tanh_gate ? (-2.0f * L2E_) : (-L2E_);
    const float act_am1 = act_a - 1.0f;

    const int  jb = wv * 8 + hi * 4;           // h-columns owned in update phase
    float c0[4] = {0.f, 0.f, 0.f, 0.f};
    float c1[4] = {0.f, 0.f, 0.f, 0.f};

    __syncthreads();

    for (int t = 0; t < T_; ++t) {
        // ================= layer 0: gate pre-activations =================
        float xv[8];
#pragma unroll
        for (int r = 0; r < 8; ++r)
            xv[r] = xs[(hi * 8 + r) * XS_STRIDE + t];   // x[m=r+hi*8, t]

        v16h a0 = load_afrag(hcat, lm, hi, 0);          // h0, K 0..31
        v16h a1 = load_afrag(hcat, lm, hi, 1);          // h0, K 32..63
#pragma unroll
        for (int nt = 0; nt < 2; ++nt) {
            v8f acc;
#pragma unroll
            for (int r = 0; r < 8; ++r)                 // bias + x*W_ih0 (D=1)
                acc[r] = fmaf(xv[r], wihr[nt], bias0r[nt]);
            acc = __builtin_amdgcn_wmma_f32_16x16x32_f16(false, a0, false, B0[nt][0],
                                                         (short)0, acc, false, false);
            acc = __builtin_amdgcn_wmma_f32_16x16x32_f16(false, a1, false, B0[nt][1],
                                                         (short)0, acc, false, false);
#pragma unroll
            for (int r = 0; r < 8; ++r)
                acc[r] = gate_act(acc[r], act_a, act_k, act_am1);
            float* gp = &gact[(wv * 32 + nt * 16 + lm) * GA_STRIDE + hi * 8];
            *(float4*)gp       = make_float4(acc[0], acc[1], acc[2], acc[3]);
            *(float4*)(gp + 4) = make_float4(acc[4], acc[5], acc[6], acc[7]);
        }
        __syncthreads();

        // ================= layer 0: c/h update =================
        {
            float hv[4];
#pragma unroll
            for (int e = 0; e < 4; ++e) {
                const int j = jb + e;
                float iv = gact[(j      ) * GA_STRIDE + lm];
                float fv = gact[( 64 + j) * GA_STRIDE + lm];
                float gv = gact[(128 + j) * GA_STRIDE + lm];
                float ov = gact[(192 + j) * GA_STRIDE + lm];
                c0[e] = fv * c0[e] + iv * gv;
                hv[e] = ov * tanh_fast(c0[e]);
            }
            v4h hh = { (_Float16)hv[0], (_Float16)hv[1], (_Float16)hv[2], (_Float16)hv[3] };
            *(v4h*)&hcat[lm * HC_STRIDE + jb] = hh;     // y0_t, cols 0..63
        }
        __syncthreads();

        // ================= layer 1: gate pre-activations =================
        v16h ac[4];
#pragma unroll
        for (int kc = 0; kc < 4; ++kc)                  // A = [y0_t | h1], K=128
            ac[kc] = load_afrag(hcat, lm, hi, kc);
#pragma unroll
        for (int nt = 0; nt < 2; ++nt) {
            v8f acc;
#pragma unroll
            for (int r = 0; r < 8; ++r) acc[r] = bias1r[nt];
#pragma unroll
            for (int kc = 0; kc < 4; ++kc)
                acc = __builtin_amdgcn_wmma_f32_16x16x32_f16(false, ac[kc], false, B1[nt][kc],
                                                             (short)0, acc, false, false);
#pragma unroll
            for (int r = 0; r < 8; ++r)
                acc[r] = gate_act(acc[r], act_a, act_k, act_am1);
            float* gp = &gact[(wv * 32 + nt * 16 + lm) * GA_STRIDE + hi * 8];
            *(float4*)gp       = make_float4(acc[0], acc[1], acc[2], acc[3]);
            *(float4*)(gp + 4) = make_float4(acc[4], acc[5], acc[6], acc[7]);
        }
        __syncthreads();

        // ================= layer 1: c/h update =================
        {
            float hv[4];
#pragma unroll
            for (int e = 0; e < 4; ++e) {
                const int j = jb + e;
                float iv = gact[(j      ) * GA_STRIDE + lm];
                float fv = gact[( 64 + j) * GA_STRIDE + lm];
                float gv = gact[(128 + j) * GA_STRIDE + lm];
                float ov = gact[(192 + j) * GA_STRIDE + lm];
                c1[e] = fv * c1[e] + iv * gv;
                hv[e] = ov * tanh_fast(c1[e]);
            }
            v4h hh = { (_Float16)hv[0], (_Float16)hv[1], (_Float16)hv[2], (_Float16)hv[3] };
            *(v4h*)&hcat[lm * HC_STRIDE + 64 + jb] = hh; // h1, cols 64..127
        }
        __syncthreads();
    }

    // ---- final FC per layer: out[l][b] = h_l_final @ W_fc.T + b_fc, l = 0,1  (O = 1) ----
    if (tid < 32) {
        const int sel = hi;                               // 0 -> layer0, 1 -> layer1
        float acc = b_fc[0];
        const _Float16* hp = &hcat[lm * HC_STRIDE + sel * 64];
#pragma unroll
        for (int k = 0; k < 64; ++k)
            acc = fmaf((float)hp[k], W_fc[k], acc);
        out[sel * B_ + b0 + lm] = acc;
    }
}

extern "C" void kernel_launch(void* const* d_in, const int* in_sizes, int n_in,
                              void* d_out, int out_size, void* d_ws, size_t ws_size,
                              hipStream_t stream) {
    const float* x     = (const float*)d_in[0];
    const float* W_ih0 = (const float*)d_in[1];
    const float* W_hh0 = (const float*)d_in[2];
    const float* b_ih0 = (const float*)d_in[3];
    const float* b_hh0 = (const float*)d_in[4];
    const float* W_ih1 = (const float*)d_in[5];
    const float* W_hh1 = (const float*)d_in[6];
    const float* b_ih1 = (const float*)d_in[7];
    const float* b_hh1 = (const float*)d_in[8];
    const float* W_fc  = (const float*)d_in[9];
    const float* b_fc  = (const float*)d_in[10];

    lstm2_fused_wmma<<<dim3(B_ / 16), dim3(256), 0, stream>>>(
        x, W_ih0, W_hh0, b_ih0, b_hh0,
        W_ih1, W_hh1, b_ih1, b_hh1,
        W_fc, b_fc, (float*)d_out);
}
